// MultiViewMamba_64682207478426
// MI455X (gfx1250) — matrix operand order repaired and checked by reading, not compiled, not run on hardware
//
#include <hip/hip_runtime.h>

#define B_ 4
#define L_ 512
#define V_ 4
#define D_ 256
#define DI_ 512
#define DS_ 16
#define DTR_ 16
#define DCONV_ 4
#define M_ (B_*L_)          // 2048 rows per view
#define E2_ (2*DI_)         // 1024
#define NX_ (DTR_+2*DS_)    // 48

typedef __attribute__((ext_vector_type(16))) _Float16 v16h;
typedef __attribute__((ext_vector_type(8)))  float    v8f;

__device__ __forceinline__ float sigm_(float x){ return 1.0f/(1.0f+__expf(-x)); }

// ---------------- weight conversion ----------------
__global__ void k_cvt(const float* __restrict__ s, _Float16* __restrict__ d, int n){
  int i = blockIdx.x*blockDim.x + threadIdx.x;
  if (i < n) d[i] = (_Float16)s[i];
}

// w_dt (V,DI,16) -> padded (V,DI,32) f16, zeros in k=16..31
__global__ void k_cvt_wdt(const float* __restrict__ s, _Float16* __restrict__ d){
  int i = blockIdx.x*blockDim.x + threadIdx.x;
  if (i >= V_*DI_*32) return;
  int k = i & 31;
  int rest = i >> 5;                    // v*DI_ + channel
  d[i] = (k < DTR_) ? (_Float16)s[rest*DTR_ + k] : (_Float16)0.0f;
}

// ---------------- layernorm (one wave per row of 256) ----------------
__global__ void k_layernorm(const float* __restrict__ x, const float* __restrict__ g,
                            const float* __restrict__ bta, _Float16* __restrict__ xv){
  int wid  = threadIdx.x >> 5;
  int lane = threadIdx.x & 31;
  int row  = blockIdx.x*(blockDim.x>>5) + wid;     // 0 .. V_*M_-1
  if (row >= V_*M_) return;
  int v = row / M_;
  int m = row - v*M_;                              // m = b*L_ + l
  const float* xp = x + ((size_t)m*V_ + v)*D_;     // x layout (B,L,V,D)
  float vals[8];
  float s = 0.f;
#pragma unroll
  for (int j=0;j<8;j++){ vals[j] = xp[lane + 32*j]; s += vals[j]; }
#pragma unroll
  for (int o=16;o>=1;o>>=1) s += __shfl_xor(s, o, 32);
  float mu = s * (1.0f/D_);
  float vs = 0.f;
#pragma unroll
  for (int j=0;j<8;j++){ float t = vals[j]-mu; vs += t*t; }
#pragma unroll
  for (int o=16;o>=1;o>>=1) vs += __shfl_xor(vs, o, 32);
  float inv = rsqrtf(vs*(1.0f/D_) + 1e-5f);
  _Float16* op = xv + ((size_t)v*M_ + m)*D_;
#pragma unroll
  for (int j=0;j<8;j++){
    int d = lane + 32*j;
    op[d] = (_Float16)((vals[j]-mu)*inv*g[v*D_+d] + bta[v*D_+d]);
  }
}

// ---------------- register-blocked WMMA GEMM: C[m,n] = sum_k A[m,k]*W[n,k] -------
// Each wave computes an (MB*16) x (NB*16) macro-tile: MB A-fragments and NB
// B-fragments are loaded per 32-wide K step and combined into MB*NB WMMAs,
// giving MB*NB/(MB+NB) fragment reuse at register level.
// mode 0: store C[v] row-major (ldc). mode 1: scatter to duplicated (B,L,V,D).
template<int MB, int NB>
__global__ void k_gemm_wmma(const _Float16* __restrict__ A, int lda, size_t aStride,
                            const _Float16* __restrict__ W, int ldw, size_t wStride,
                            float* __restrict__ C, int ldc, size_t cStride,
                            int MBlocks, int NBlocks, int K, int mode){
  int v    = blockIdx.z;
  int wpb  = blockDim.x >> 5;
  int wid  = threadIdx.x >> 5;
  int blk  = blockIdx.x*wpb + wid;
  if (blk >= MBlocks*NBlocks) return;
  int bm = blk / NBlocks, bn = blk - bm*NBlocks;
  int lane = threadIdx.x & 31;
  int hh   = lane >> 4;            // half-wave select
  int r    = lane & 15;            // A: M row / B,C: N column
  const _Float16* Abase = A + v*aStride + (size_t)(bm*MB*16 + r)*lda;
  const _Float16* Wbase = W + v*wStride + (size_t)(bn*NB*16 + r)*ldw;
  v8f acc[MB][NB];
#pragma unroll
  for (int mb=0;mb<MB;mb++)
#pragma unroll
    for (int nb=0;nb<NB;nb++) acc[mb][nb] = {};

  for (int k0 = 0; k0 < K; k0 += 32){
    v16h a[MB], bf[NB];
    // A 16x32 f16 fragment (ISA 7.12.2): lane half -> K+8; VGPRs 0-3 = K 0..7, 4-7 = K 16..23
#pragma unroll
    for (int mb=0;mb<MB;mb++){
      const _Float16* ap = Abase + (size_t)mb*16*lda + k0 + 8*hh;
#pragma unroll
      for (int t=0;t<8;t++){ a[mb][t] = ap[t]; a[mb][8+t] = ap[16+t]; }
    }
    // B 32x16 f16 fragment: lane = N col, same K order as A (lane half -> K+8; elems 0-7 = K 0..7, 8-15 = K 16..23)
#pragma unroll
    for (int nb=0;nb<NB;nb++){
      const _Float16* bp = Wbase + (size_t)nb*16*ldw + k0 + 8*hh;
#pragma unroll
      for (int t=0;t<8;t++){ bf[nb][t] = bp[t]; bf[nb][8+t] = bp[16+t]; }
    }
#pragma unroll
    for (int mb=0;mb<MB;mb++)
#pragma unroll
      for (int nb=0;nb<NB;nb++)
        acc[mb][nb] = __builtin_amdgcn_wmma_f32_16x16x32_f16(
            false, a[mb], false, bf[nb], (short)0, acc[mb][nb], false, false);
  }

  if (mode == 0){
    float* cp = C + v*cStride;
#pragma unroll
    for (int mb=0;mb<MB;mb++)
#pragma unroll
      for (int i=0;i<8;i++){
        int m = bm*MB*16 + mb*16 + i + 8*hh;
#pragma unroll
        for (int nb=0;nb<NB;nb++){
          int n = bn*NB*16 + nb*16 + r;
          cp[(size_t)m*ldc + n] = acc[mb][nb][i];
        }
      }
  } else {
    const size_t TOT = (size_t)B_*L_*V_*D_;
#pragma unroll
    for (int mb=0;mb<MB;mb++)
#pragma unroll
      for (int i=0;i<8;i++){
        int m = bm*MB*16 + mb*16 + i + 8*hh;          // m = b*L_ + l
#pragma unroll
        for (int nb=0;nb<NB;nb++){
          int n = bn*NB*16 + nb*16 + r;
          size_t idx = ((size_t)m*V_ + v)*D_ + n;     // (B,L,V,D)
          C[idx]       = acc[mb][nb][i];
          C[idx + TOT] = acc[mb][nb][i];
        }
      }
  }
}

// ---------------- causal depthwise conv(4) + SiLU ----------------
__global__ void k_conv(const float* __restrict__ xz, const float* __restrict__ cw,
                       const float* __restrict__ cb, float* __restrict__ xc,
                       _Float16* __restrict__ xch){
  size_t i = (size_t)blockIdx.x*blockDim.x + threadIdx.x;
  if (i >= (size_t)V_*M_*DI_) return;
  int ch = (int)(i % DI_);
  int m  = (int)((i / DI_) % M_);
  int v  = (int)(i / ((size_t)DI_*M_));
  int l  = m & (L_-1);
  const float* wv = cw + ((size_t)v*DI_ + ch)*DCONV_;
  float acc = cb[v*DI_ + ch];
#pragma unroll
  for (int j=0;j<DCONV_;j++){
    int lp = l + j - (DCONV_-1);
    if (lp >= 0)
      acc += xz[((size_t)v*M_ + (m + j - (DCONV_-1)))*E2_ + ch] * wv[j];
  }
  float s = acc * sigm_(acc);
  xc[i]  = s;
  xch[i] = (_Float16)s;
}

// ---------------- extract dt_lo, pad K 16->32, f16 ----------------
__global__ void k_dtlo(const float* __restrict__ xdbl, _Float16* __restrict__ dtlo){
  int i = blockIdx.x*blockDim.x + threadIdx.x;
  if (i >= V_*M_*32) return;
  int k = i & 31;
  int m = (i >> 5) % M_;
  int v = i / (32*M_);
  float val = (k < DTR_) ? xdbl[((size_t)v*M_ + m)*NX_ + k] : 0.0f;
  dtlo[i] = (_Float16)val;
}

// ---------------- selective scan: state in registers, B/C via LDS ----------------
__global__ void k_scan(const float* __restrict__ dlin, const float* __restrict__ bdt,
                       const float* __restrict__ Alog, const float* __restrict__ Dsk,
                       const float* __restrict__ xdbl, const float* __restrict__ xc,
                       const float* __restrict__ xz, _Float16* __restrict__ yh){
  int b = blockIdx.x, v = blockIdx.y;
  int t = threadIdx.x;                 // 256 threads, 2 channels each
  __shared__ float Bs[DS_], Cs[DS_];
  int d0 = t*2;
  float Areg[2][DS_], h[2][DS_];
#pragma unroll
  for (int c=0;c<2;c++)
#pragma unroll
    for (int n=0;n<DS_;n++){
      Areg[c][n] = -__expf(Alog[((size_t)v*DI_ + d0 + c)*DS_ + n]);
      h[c][n] = 0.f;
    }
  float bd[2] = { bdt[v*DI_+d0], bdt[v*DI_+d0+1] };
  float dk[2] = { Dsk[v*DI_+d0], Dsk[v*DI_+d0+1] };
  for (int l=0;l<L_;l++){
    int m = b*L_ + l;
    const size_t rowx = ((size_t)v*M_ + m);
    if (t < DS_)          Bs[t]      = xdbl[rowx*NX_ + DTR_ + t];
    else if (t < 2*DS_)   Cs[t-DS_]  = xdbl[rowx*NX_ + DTR_ + DS_ + (t-DS_)];
    __syncthreads();
#pragma unroll
    for (int c=0;c<2;c++){
      int d = d0 + c;
      float dl    = dlin[rowx*DI_ + d] + bd[c];
      float delta = (dl > 20.f) ? dl : log1pf(__expf(dl));     // softplus
      float xcv   = xc[rowx*DI_ + d];
      float dx    = delta * xcv;
      float y = 0.f;
#pragma unroll
      for (int n=0;n<DS_;n++){
        float hv = __expf(delta * Areg[c][n]) * h[c][n] + dx * Bs[n];
        h[c][n] = hv;
        y += hv * Cs[n];
      }
      float z  = xz[rowx*E2_ + DI_ + d];
      float yf = (y + xcv*dk[c]) * (z * sigm_(z));              // + D skip, * silu(z)
      yh[rowx*DI_ + d] = (_Float16)yf;
    }
    __syncthreads();
  }
}

extern "C" void kernel_launch(void* const* d_in, const int* in_sizes, int n_in,
                              void* d_out, int out_size, void* d_ws, size_t ws_size,
                              hipStream_t stream){
  const float* x     = (const float*)d_in[0];
  const float* ln_g  = (const float*)d_in[1];
  const float* ln_b  = (const float*)d_in[2];
  const float* w_in  = (const float*)d_in[3];
  const float* convw = (const float*)d_in[4];
  const float* convb = (const float*)d_in[5];
  const float* w_x   = (const float*)d_in[6];
  const float* w_dt  = (const float*)d_in[7];
  const float* b_dt  = (const float*)d_in[8];
  const float* A_log = (const float*)d_in[9];
  const float* D_sk  = (const float*)d_in[10];
  const float* w_out = (const float*)d_in[11];
  float* out = (float*)d_out;

  char* ws = (char*)d_ws;
  size_t off = 0;
  auto alloc = [&](size_t bytes){ void* p = ws + off; off += (bytes + 255) & ~(size_t)255; return p; };

  _Float16* w_in_h  = (_Float16*)alloc((size_t)V_*E2_*D_   *2);
  _Float16* w_x_h   = (_Float16*)alloc((size_t)V_*NX_*DI_  *2);
  _Float16* w_dt_h  = (_Float16*)alloc((size_t)V_*DI_*32   *2);
  _Float16* w_out_h = (_Float16*)alloc((size_t)V_*D_*DI_   *2);
  _Float16* xv_h    = (_Float16*)alloc((size_t)V_*M_*D_    *2);
  float*    xz      = (float*)   alloc((size_t)V_*M_*E2_   *4);
  float*    xc      = (float*)   alloc((size_t)V_*M_*DI_   *4);
  _Float16* xc_h    = (_Float16*)alloc((size_t)V_*M_*DI_   *2);
  float*    xdbl    = (float*)   alloc((size_t)V_*M_*NX_   *4);
  _Float16* dtlo_h  = (_Float16*)alloc((size_t)V_*M_*32    *2);
  float*    dlin    = (float*)   alloc((size_t)V_*M_*DI_   *4);
  _Float16* y_h     = (_Float16*)alloc((size_t)V_*M_*DI_   *2);

  // weights -> f16
  { int n = V_*E2_*D_;  k_cvt<<<(n+255)/256,256,0,stream>>>(w_in,  w_in_h,  n); }
  { int n = V_*NX_*DI_; k_cvt<<<(n+255)/256,256,0,stream>>>(w_x,   w_x_h,   n); }
  { int n = V_*D_*DI_;  k_cvt<<<(n+255)/256,256,0,stream>>>(w_out, w_out_h, n); }
  { int n = V_*DI_*32;  k_cvt_wdt<<<(n+255)/256,256,0,stream>>>(w_dt, w_dt_h); }

  // layernorm -> xv_h
  k_layernorm<<<(V_*M_)/8, 256, 0, stream>>>(x, ln_g, ln_b, xv_h);

  // G1: xz = xv @ w_in^T   (M=2048, N=1024, K=256), 32x64 macro-tiles
  { int blks = (M_/32)*(E2_/64);     // 64 * 16 = 1024 waves/view
    k_gemm_wmma<2,4><<<dim3((blks+3)/4,1,V_),128,0,stream>>>(
      xv_h, D_, (size_t)M_*D_, w_in_h, D_, (size_t)E2_*D_,
      xz, E2_, (size_t)M_*E2_, M_/32, E2_/64, D_, 0); }

  // depthwise conv + silu
  { size_t n = (size_t)V_*M_*DI_;
    k_conv<<<(unsigned)((n+255)/256),256,0,stream>>>(xz, convw, convb, xc, xc_h); }

  // G2: xdbl = xc @ w_x^T  (M=2048, N=48, K=512), 64x48 macro-tiles
  { int blks = (M_/64)*1;            // 32 waves/view
    k_gemm_wmma<4,3><<<dim3((blks+3)/4,1,V_),128,0,stream>>>(
      xc_h, DI_, (size_t)M_*DI_, w_x_h, DI_, (size_t)NX_*DI_,
      xdbl, NX_, (size_t)M_*NX_, M_/64, 1, DI_, 0); }

  // dt_lo padded f16
  { int n = V_*M_*32; k_dtlo<<<(n+255)/256,256,0,stream>>>(xdbl, dtlo_h); }

  // G3: dlin = dt_lo @ w_dt^T  (M=2048, N=512, K=32 padded), 32x64 macro-tiles
  { int blks = (M_/32)*(DI_/64);     // 64 * 8 = 512 waves/view
    k_gemm_wmma<2,4><<<dim3((blks+3)/4,1,V_),128,0,stream>>>(
      dtlo_h, 32, (size_t)M_*32, w_dt_h, 32, (size_t)DI_*32,
      dlin, DI_, (size_t)M_*DI_, M_/32, DI_/64, 32, 0); }

  // selective scan + gating -> y_h
  k_scan<<<dim3(B_,V_),256,0,stream>>>(dlin, b_dt, A_log, D_sk, xdbl, xc, xz, y_h);

  // G4: out = y @ w_out^T (M=2048, N=256, K=512), scattered to duplicated output
  { int blks = (M_/32)*(D_/64);      // 64 * 4 = 256 waves/view
    k_gemm_wmma<2,4><<<dim3((blks+3)/4,1,V_),128,0,stream>>>(
      y_h, DI_, (size_t)M_*DI_, w_out_h, DI_, (size_t)D_*DI_,
      out, 0, 0, M_/32, D_/64, DI_, 1); }
}
